// PowerSoftmaxAttention_47691316855187
// MI455X (gfx1250) — matrix operand order repaired and checked by reading, not compiled, not run on hardware
//
#include <hip/hip_runtime.h>
#include <hip/hip_bf16.h>

typedef __attribute__((ext_vector_type(16))) _Float16 v16h;
typedef __attribute__((ext_vector_type(8)))  _Float16 v8h;
typedef __attribute__((ext_vector_type(8)))  float    v8f;

namespace {
constexpr int   S_LEN  = 2048;
constexpr int   HD     = 64;
constexpr int   SPITCH = 2052;   // 2048 + 4 floats padding (LDS bank decorrelation)
constexpr int   VTP    = 40;     // 32 + 8 halves padding for V^T tile rows
constexpr int   WAVES  = 2;      // waves per workgroup (split-K cooperators)
constexpr float SCALE  = 0.125f; // 1/sqrt(64)
constexpr float EPS    = 1e-6f;
}

// One 2-wave workgroup owns a 16-row Q tile of one (b,h); the waves split the
// 2048-long kv sweep. ~146 KB LDS per workgroup -> 2 workgroups (4 waves) per
// 320 KB WGP, filling all four SIMD32s.
//   Phase A: scores^2 for 16x2048 into LDS via v_wmma_f32_16x16x32_f16
//            (waves interleave N-tiles), per-wave partial row sums.
//   Phase B: shfl_xor tree + cross-wave LDS combine -> 1/(den).
//   Phase C1: normalized prob, coalesced float4 stores (rows split 8/8).
//   Phase C2: out = prob @ V with WMMA; waves interleave 32-row V tiles with
//             private LDS V^T staging (no inner barriers), 4 KB LDS reduce.
__launch_bounds__(32 * WAVES)
__global__ void pow2_attn_wmma_kernel(const float* __restrict__ Qg,
                                      const float* __restrict__ Kg,
                                      const float* __restrict__ Vg,
                                      float* __restrict__ Og,
                                      float* __restrict__ Pg) {
  __shared__ float    num_lds[16 * SPITCH];       // unnormalized scores^2, fp32
  __shared__ _Float16 vt_lds[WAVES][HD * VTP];    // per-wave V^T tile, fp16
  __shared__ float    sum_lds[WAVES][16];         // per-wave partial row sums
  __shared__ float    inv_lds[16];                // per-row 1/(den)
  __shared__ float    red_lds[16 * HD];           // out-tile cross-wave reduce

  const int tid  = threadIdx.x;
  const int wave = tid >> 5;    // 0..WAVES-1
  const int lane = tid & 31;
  const int half = lane >> 4;   // 0: lanes 0-15, 1: lanes 16-31
  const int l16  = lane & 15;
  const int bh   = blockIdx.x >> 7;        // 128 q-tiles per (b,h)
  const int q0   = (blockIdx.x & 127) << 4;

  const float* Q = Qg + (size_t)bh * S_LEN * HD;
  const float* K = Kg + (size_t)bh * S_LEN * HD;
  const float* V = Vg + (size_t)bh * S_LEN * HD;
  float*       O = Og + (size_t)bh * S_LEN * HD;
  float*       P = Pg + (size_t)bh * S_LEN * S_LEN;

  // ---------------- Q A-fragments (pre-scaled, f16), one per 32-wide K-dim half.
  // A k order used here (contiguous, same as B): lane<16 -> row=l16, K in 0..15;
  //                     lane>=16 -> row=l16, K in 16..31.
  v16h qa[2];
  {
    const float* qrow = Q + (size_t)(q0 + l16) * HD;
    const int b1 = half * 16;
#pragma unroll
    for (int h = 0; h < 2; ++h) {
      const float* p = qrow + h * 32;
      float4 c0 = *(const float4*)(p + b1 + 0);
      float4 c1 = *(const float4*)(p + b1 + 4);
      float4 c2 = *(const float4*)(p + b1 + 8);
      float4 c3 = *(const float4*)(p + b1 + 12);
      float t[16] = {c0.x, c0.y, c0.z, c0.w, c1.x, c1.y, c1.z, c1.w,
                     c2.x, c2.y, c2.z, c2.w, c3.x, c3.y, c3.z, c3.w};
#pragma unroll
      for (int i = 0; i < 16; ++i) qa[h][i] = (_Float16)(t[i] * SCALE);
    }
  }

  // ---------------- Phase A: QK^T sweep (waves interleave N-tiles).
  float rs[8];
#pragma unroll
  for (int r = 0; r < 8; ++r) rs[r] = 0.0f;

  for (int i = 0; i < S_LEN / 16 / WAVES; ++i) {
    const int nt = i * WAVES + wave;
    const int kvrow = nt * 16 + l16;
    v8f c = {};
#pragma unroll
    for (int h = 0; h < 2; ++h) {
      // B k order used here (contiguous, same as A): lane<16 -> col=l16, K=0..15; lane>=16 -> K=16..31.
      // For QK^T, B[kk][n] = K[n][kk]: 16 contiguous f32 from K row `kvrow`.
      const float* p = K + (size_t)kvrow * HD + h * 32 + half * 16;
      if (h == 0) __builtin_prefetch(p + WAVES * 16 * HD, 0, 1);  // next tile for this wave
      float4 c0 = *(const float4*)(p + 0);
      float4 c1 = *(const float4*)(p + 4);
      float4 c2 = *(const float4*)(p + 8);
      float4 c3 = *(const float4*)(p + 12);
      float t[16] = {c0.x, c0.y, c0.z, c0.w, c1.x, c1.y, c1.z, c1.w,
                     c2.x, c2.y, c2.z, c2.w, c3.x, c3.y, c3.z, c3.w};
      v16h kb;
#pragma unroll
      for (int j = 0; j < 16; ++j) kb[j] = (_Float16)t[j];
      c = __builtin_amdgcn_wmma_f32_16x16x32_f16(false, qa[h], false, kb,
                                                 (short)0, c, false, false);
    }
    // C layout: VGPR r, lane -> M = r + 8*half, N = l16.
    float* drow = num_lds + (8 * half) * SPITCH + nt * 16 + l16;
#pragma unroll
    for (int r = 0; r < 8; ++r) {
      const float n2 = c[r] * c[r];
      drow[r * SPITCH] = n2;
      rs[r] += n2;
    }
  }

  // ---------------- Phase B: intra-wave shfl tree, then cross-wave combine.
#pragma unroll
  for (int r = 0; r < 8; ++r) {
    float s = rs[r];
    s += __shfl_xor(s, 1);
    s += __shfl_xor(s, 2);
    s += __shfl_xor(s, 4);
    s += __shfl_xor(s, 8);
    rs[r] = s;
  }
  if (l16 == 0) {
#pragma unroll
    for (int r = 0; r < 8; ++r) sum_lds[wave][r + 8 * half] = rs[r];
  }
  __syncthreads();
  if (wave == 0 && l16 == 0) {
#pragma unroll
    for (int r = 0; r < 8; ++r) {
      const int m = r + 8 * half;
      float s = EPS;
#pragma unroll
      for (int w = 0; w < WAVES; ++w) s += sum_lds[w][m];
      inv_lds[m] = 1.0f / s;
    }
  }
  __syncthreads();

  // ---------------- Phase C1: normalized prob, coalesced float4 stores.
  for (int rr = 0; rr < 16 / WAVES; ++rr) {
    const int r = wave * (16 / WAVES) + rr;
    const float inv = inv_lds[r];
    const float* nrow = num_lds + r * SPITCH;
    float* prow = P + (size_t)(q0 + r) * S_LEN;
    for (int c0 = lane * 4; c0 < S_LEN; c0 += 128) {
      float4 t = *(const float4*)(nrow + c0);
      t.x *= inv; t.y *= inv; t.z *= inv; t.w *= inv;
      *(float4*)(prow + c0) = t;
    }
  }

  // ---------------- Phase C2: out = prob @ V via WMMA (waves interleave k-tiles).
  v8f oc[4] = {{}, {}, {}, {}};
  const float invm = inv_lds[l16];
  const float* myrow = num_lds + l16 * SPITCH;
  _Float16* vt = vt_lds[wave];

  for (int i = 0; i < S_LEN / 32 / WAVES; ++i) {
    const int kv0 = (i * WAVES + wave) * 32;

    // Stage V tile transposed into this wave's LDS buffer: vt[d][kk].
    {
      const float* vr = V + (size_t)(kv0 + lane) * HD;
#pragma unroll
      for (int d0 = 0; d0 < HD; d0 += 4) {
        float4 t = *(const float4*)(vr + d0);
        vt[(d0 + 0) * VTP + lane] = (_Float16)t.x;
        vt[(d0 + 1) * VTP + lane] = (_Float16)t.y;
        vt[(d0 + 2) * VTP + lane] = (_Float16)t.z;
        vt[(d0 + 3) * VTP + lane] = (_Float16)t.w;
      }
    }
    // Buffer is wave-private: in-wave DS ordering suffices, no barrier.

    // A fragment: normalized prob row l16, contiguous kv chunks, k = 16*half + s (matches B).
    v16h pa;
    {
      const int b1 = kv0 + half * 16;
      float4 a0 = *(const float4*)(myrow + b1 + 0);
      float4 a1 = *(const float4*)(myrow + b1 + 4);
      float4 a2 = *(const float4*)(myrow + b1 + 8);
      float4 a3 = *(const float4*)(myrow + b1 + 12);
      float t[16] = {a0.x, a0.y, a0.z, a0.w, a1.x, a1.y, a1.z, a1.w,
                     a2.x, a2.y, a2.z, a2.w, a3.x, a3.y, a3.z, a3.w};
#pragma unroll
      for (int j = 0; j < 16; ++j) pa[j] = (_Float16)(t[j] * invm);
    }

#pragma unroll
    for (int nt = 0; nt < 4; ++nt) {
      // B fragment from V^T tile: lane -> col n = nt*16 + l16, 16 contiguous kk.
      const _Float16* vrow = vt + (nt * 16 + l16) * VTP + half * 16;
      v8h lo = *(const v8h*)(vrow + 0);
      v8h hi = *(const v8h*)(vrow + 8);
      v16h vb;
#pragma unroll
      for (int j = 0; j < 8; ++j) { vb[j] = lo[j]; vb[j + 8] = hi[j]; }
      oc[nt] = __builtin_amdgcn_wmma_f32_16x16x32_f16(false, pa, false, vb,
                                                      (short)0, oc[nt], false, false);
    }
  }

  // ---------------- Epilogue: cross-wave reduce, then write out tile.
  // C layout: lane -> M = r + 8*half, N = l16 (+ nt*16).
  if (wave == 1) {
#pragma unroll
    for (int nt = 0; nt < 4; ++nt)
#pragma unroll
      for (int r = 0; r < 8; ++r)
        red_lds[(r + 8 * half) * HD + nt * 16 + l16] = oc[nt][r];
  }
  __syncthreads();
  if (wave == 0) {
#pragma unroll
    for (int nt = 0; nt < 4; ++nt) {
#pragma unroll
      for (int r = 0; r < 8; ++r) {
        const int m = r + 8 * half;
        const int n = nt * 16 + l16;
        O[(size_t)(q0 + m) * HD + n] = oc[nt][r] + red_lds[m * HD + n];
      }
    }
  }
}

extern "C" void kernel_launch(void* const* d_in, const int* in_sizes, int n_in,
                              void* d_out, int out_size, void* d_ws, size_t ws_size,
                              hipStream_t stream) {
  (void)in_sizes; (void)n_in; (void)d_ws; (void)ws_size; (void)out_size;
  const float* q = (const float*)d_in[0];
  const float* k = (const float*)d_in[1];
  const float* v = (const float*)d_in[2];

  // d_out = concat(out [2*16*2048*64], prob [2*16*2048*2048]), both f32.
  float* out  = (float*)d_out;
  float* prob = out + (size_t)2 * 16 * 2048 * 64;

  const int n_bh    = 2 * 16;
  const int n_qtile = S_LEN / 16;            // 128
  dim3 grid(n_bh * n_qtile);                 // 4096 workgroups, 2 waves each
  dim3 block(32 * WAVES);
  pow2_attn_wmma_kernel<<<grid, block, 0, stream>>>(q, k, v, out, prob);
}
